// MaskedSelfAttention_70420283785308
// MI455X (gfx1250) — matrix operand order repaired and checked by reading, not compiled, not run on hardware
//
#include <hip/hip_runtime.h>
#include <hip/hip_bf16.h>

// MI455X / gfx1250, wave32, WMMA f16->f32, async-to-LDS tile staging.
//
// Pipeline:
//   1) cvt_f32_f16: hidden, Wq, Wk, Wv -> f16 workspace
//   2) gemm_wmma (x3): Q/K/V = X @ W^T + b   (double-buffered async LDS staging,
//      pipeline unrolled x2 so buffer selection is static)
//   3) attn_flash:  per (b,h) flash attention with mask + rel bias -> ctx f32
//   4) gemm_wmma (batched): mean_scores = (Qfull @ Kfull^T)/128 -> f32
//      (head-mean of raw scores == full-D dot / (H*sqrt(DH)))

typedef __attribute__((ext_vector_type(16))) _Float16 v16h;
typedef __attribute__((ext_vector_type(8)))  _Float16 v8h;
typedef __attribute__((ext_vector_type(8)))  float    v8f;

#define BB 4
#define SS 2048
#define DD 1024
#define HH 16
#define DH 64

static __device__ __forceinline__ v8f wmma_f16(v16h a, v16h b, v8f c) {
    return __builtin_amdgcn_wmma_f32_16x16x32_f16(
        /*neg_a=*/false, a, /*neg_b=*/false, b,
        /*c_mod=*/(short)0, c, /*reuse_a=*/false, /*reuse_b=*/false);
}

// A-fragment (16x32 f16, MxK): lane ln holds row M=ln; element i: k = g*16 + hi*8 + j
static __device__ __forceinline__ v16h load_afrag(const _Float16* base, int stride,
                                                  int ln, int hi) {
    const _Float16* p = base + (size_t)ln * stride + hi * 8;
    v8h c0 = *(const v8h*)p;
    v8h c1 = *(const v8h*)(p + 16);
    return __builtin_shufflevector(c0, c1, 0,1,2,3,4,5,6,7,8,9,10,11,12,13,14,15);
}

// B-fragment (32x16 f16, KxN) from row-major [N][K] storage (i.e. computing A@B^T):
// lane ln holds column N=ln; element i: k = g*16 + hi*8 + j  (same as A-fragment)
static __device__ __forceinline__ v16h load_bfrag(const _Float16* base, int stride,
                                                  int ln, int hi) {
    const _Float16* p = base + (size_t)ln * stride + hi * 8;
    v8h c0 = *(const v8h*)p;
    v8h c1 = *(const v8h*)(p + 16);
    return __builtin_shufflevector(c0, c1, 0,1,2,3,4,5,6,7,8,9,10,11,12,13,14,15);
}

// Async copy 32 bytes (16 halves) global -> LDS, tracked by ASYNCcnt.
// LDS byte address = low 32 bits of the generic pointer (LDS aperture).
static __device__ __forceinline__ void async_copy_16h(const _Float16* g, _Float16* l) {
    unsigned lo = (unsigned)(uintptr_t)l;
    unsigned long long ga = (unsigned long long)(uintptr_t)g;
    asm volatile("global_load_async_to_lds_b128 %0, %1, off"
                 :: "v"(lo), "v"(ga) : "memory");
    asm volatile("global_load_async_to_lds_b128 %0, %1, off"
                 :: "v"(lo + 16u), "v"(ga + 16ull) : "memory");
}

static __device__ __forceinline__ void wait_async0() {
    asm volatile("s_wait_asynccnt 0" ::: "memory");
}

__global__ void cvt_f32_f16(const float* __restrict__ src,
                            _Float16* __restrict__ dst, int n) {
    int i = blockIdx.x * blockDim.x + threadIdx.x;
    if (i < n) dst[i] = (_Float16)__builtin_nontemporal_load(src + i);
}

// C[M,N] = scale * (A[M,K] @ Bm[N,K]^T) + bias[N]
// Block tile 128(M) x 128(N), K-step 32, double-buffered async LDS staging.
// 256 threads = 8 waves in 4(M)x2(N); each wave owns a 32x64 sub-tile (2x4 frags).
// K/32 must be even (K=1024 here) -> pipeline unrolled x2, static buffers.
__global__ __launch_bounds__(256) void gemm_wmma(
    const _Float16* __restrict__ A, const _Float16* __restrict__ Bm,
    const float* __restrict__ bias,
    _Float16* __restrict__ outH, float* __restrict__ outF,
    int M, int N, int K, float scale,
    long strideA, long strideB, long strideC) {
    A  += (size_t)blockIdx.z * strideA;
    Bm += (size_t)blockIdx.z * strideB;

    const int tid = threadIdx.x;
    const int wave = tid >> 5, lane = tid & 31;
    const int hi = lane >> 4, ln = lane & 15;
    const int wm = wave >> 1, wn = wave & 1;
    const int m0 = blockIdx.x * 128, n0 = blockIdx.y * 128;

    __shared__ alignas(32) _Float16 As[2][128 * 32];
    __shared__ alignas(32) _Float16 Bs[2][128 * 32];

    // staging: thread t copies 16 halves of row (t>>1), half-row (t&1)
    const int srow = tid >> 1, spart = tid & 1;
    const _Float16* gA = A + (size_t)(m0 + srow) * K + spart * 16;
    const _Float16* gB = Bm + (size_t)(n0 + srow) * K + spart * 16;
    _Float16* lA0 = &As[0][srow * 32 + spart * 16];
    _Float16* lB0 = &Bs[0][srow * 32 + spart * 16];
    _Float16* lA1 = &As[1][srow * 32 + spart * 16];
    _Float16* lB1 = &Bs[1][srow * 32 + spart * 16];

    v8f acc[2][4] = {};

    auto compute = [&](const _Float16* as, const _Float16* bs) {
        v16h af[2], bf[4];
#pragma unroll
        for (int mt = 0; mt < 2; ++mt)
            af[mt] = load_afrag(&as[(wm * 32 + mt * 16) * 32], 32, ln, hi);
#pragma unroll
        for (int nt = 0; nt < 4; ++nt)
            bf[nt] = load_bfrag(&bs[(wn * 64 + nt * 16) * 32], 32, ln, hi);
#pragma unroll
        for (int mt = 0; mt < 2; ++mt)
#pragma unroll
            for (int nt = 0; nt < 4; ++nt)
                acc[mt][nt] = wmma_f16(af[mt], bf[nt], acc[mt][nt]);
    };

    const int nk = K >> 5;           // even by construction
    async_copy_16h(gA, lA0);
    async_copy_16h(gB, lB0);

    for (int i = 0; i < nk; i += 2) {
        wait_async0();
        __syncthreads();
        // i+1 < nk always holds (nk even)
        async_copy_16h(gA + (size_t)(i + 1) * 32, lA1);
        async_copy_16h(gB + (size_t)(i + 1) * 32, lB1);
        compute(As[0], Bs[0]);

        wait_async0();
        __syncthreads();
        if (i + 2 < nk) {
            async_copy_16h(gA + (size_t)(i + 2) * 32, lA0);
            async_copy_16h(gB + (size_t)(i + 2) * 32, lB0);
        }
        compute(As[1], Bs[1]);
    }

    // epilogue: hoist bias, branch once on output format
    float bv[4];
    int gcol[4];
#pragma unroll
    for (int nt = 0; nt < 4; ++nt) {
        gcol[nt] = n0 + wn * 64 + nt * 16 + ln;
        bv[nt] = bias ? bias[gcol[nt]] : 0.0f;
    }
    if (outH) {
        _Float16* out = outH + (size_t)blockIdx.z * strideC;
#pragma unroll
        for (int mt = 0; mt < 2; ++mt)
#pragma unroll
            for (int nt = 0; nt < 4; ++nt)
#pragma unroll
                for (int r = 0; r < 8; ++r) {
                    int grow = m0 + wm * 32 + mt * 16 + hi * 8 + r;
                    out[(size_t)grow * N + gcol[nt]] =
                        (_Float16)(acc[mt][nt][r] * scale + bv[nt]);
                }
    } else {
        // f32 path = mean-scores output: write-once, never re-read -> non-temporal
        float* out = outF + (size_t)blockIdx.z * strideC;
#pragma unroll
        for (int mt = 0; mt < 2; ++mt)
#pragma unroll
            for (int nt = 0; nt < 4; ++nt)
#pragma unroll
                for (int r = 0; r < 8; ++r) {
                    int grow = m0 + wm * 32 + mt * 16 + hi * 8 + r;
                    __builtin_nontemporal_store(acc[mt][nt][r] * scale + bv[nt],
                                                out + (size_t)grow * N + gcol[nt]);
                }
    }
}

// Flash attention: grid (S/128, H, B); 8 waves, each wave = 16 q rows.
// K tile staged via async-to-LDS; V staged transposed (load + scatter store).
__global__ __launch_bounds__(256) void attn_flash(
    const _Float16* __restrict__ Q, const _Float16* __restrict__ Kg,
    const _Float16* __restrict__ Vg,
    const float* __restrict__ rel, const int* __restrict__ mask,
    float* __restrict__ ctx) {
    const int tid = threadIdx.x;
    const int wave = tid >> 5, lane = tid & 31;
    const int hi = lane >> 4, ln = lane & 15;
    const int b = blockIdx.z, h = blockIdx.y;
    const int q0 = blockIdx.x * 128 + wave * 16;

    __shared__ alignas(32) _Float16 Kl[64 * 64];
    __shared__ alignas(32) _Float16 Vt[64 * 64];   // transposed: [dh][key]
    __shared__ alignas(32) _Float16 Pl[8][16 * 64];

    // Q A-fragments (row = q0 + ln, dh split into two K=32 chunks)
    v16h aq[2];
    {
        const _Float16* qp = Q + ((size_t)(b * SS + q0 + ln)) * DD + h * DH;
#pragma unroll
        for (int f = 0; f < 2; ++f) {
            v8h c0 = *(const v8h*)(qp + f * 32 + hi * 8);
            v8h c1 = *(const v8h*)(qp + f * 32 + 16 + hi * 8);
            aq[f] = __builtin_shufflevector(c0, c1, 0,1,2,3,4,5,6,7,8,9,10,11,12,13,14,15);
        }
    }

    v8f o[4] = {};
    float mrun[8], lrun[8];
#pragma unroll
    for (int r = 0; r < 8; ++r) { mrun[r] = -3e30f; lrun[r] = 0.f; }

    const int skey = tid >> 2, sc = (tid & 3) * 16;

    for (int kb = 0; kb < SS; kb += 64) {
        __syncthreads();
        {   // K tile: async global->LDS (row-major)
            const _Float16* gk = Kg + ((size_t)(b * SS + kb + skey)) * DD + h * DH + sc;
            async_copy_16h(gk, &Kl[skey * 64 + sc]);
            // V tile: load + transpose-scatter into LDS
            const _Float16* gv = Vg + ((size_t)(b * SS + kb + skey)) * DD + h * DH + sc;
            v8h v0 = *(const v8h*)gv, v1 = *(const v8h*)(gv + 8);
#pragma unroll
            for (int j = 0; j < 8; ++j) {
                Vt[(sc + j) * 64 + skey]     = v0[j];
                Vt[(sc + 8 + j) * 64 + skey] = v1[j];
            }
        }
        wait_async0();
        __syncthreads();

        // S = Q @ K^T  (4 n-tiles of 16 keys; two K=32 WMMAs each)
        v8f s[4];
#pragma unroll
        for (int t = 0; t < 4; ++t) {
            v8f z = {};
            z = wmma_f16(aq[0], load_bfrag(&Kl[(t * 16) * 64], 64, ln, hi), z);
            z = wmma_f16(aq[1], load_bfrag(&Kl[(t * 16) * 64 + 32], 64, ln, hi), z);
            s[t] = z;
        }

        // scale, mask, rel bias (C layout: m = hi*8+r, n = ln)
        // mask/rel are reused across the 16 heads -> keep regular-temporal loads
#pragma unroll
        for (int t = 0; t < 4; ++t)
#pragma unroll
            for (int r = 0; r < 8; ++r) {
                int qrow = q0 + hi * 8 + r;
                int key  = kb + t * 16 + ln;
                size_t idx = ((size_t)b * SS + qrow) * SS + key;
                float sv = s[t][r] * 0.125f;
                sv = (mask[idx] == 0 ? -1e9f : sv) + rel[idx];
                s[t][r] = sv;
            }

        // online softmax
        float mloc[8], lsum[8], alpha[8];
#pragma unroll
        for (int r = 0; r < 8; ++r)
            mloc[r] = fmaxf(fmaxf(s[0][r], s[1][r]), fmaxf(s[2][r], s[3][r]));
#pragma unroll
        for (int off = 1; off < 16; off <<= 1)
#pragma unroll
            for (int r = 0; r < 8; ++r)
                mloc[r] = fmaxf(mloc[r], __shfl_xor(mloc[r], off, 16));
#pragma unroll
        for (int r = 0; r < 8; ++r) {
            float mn = fmaxf(mrun[r], mloc[r]);
            alpha[r] = __expf(mrun[r] - mn);
            mrun[r] = mn;
        }
#pragma unroll
        for (int t = 0; t < 4; ++t)
#pragma unroll
            for (int r = 0; r < 8; ++r)
                s[t][r] = __expf(s[t][r] - mrun[r]);
#pragma unroll
        for (int r = 0; r < 8; ++r)
            lsum[r] = s[0][r] + s[1][r] + s[2][r] + s[3][r];
#pragma unroll
        for (int off = 1; off < 16; off <<= 1)
#pragma unroll
            for (int r = 0; r < 8; ++r)
                lsum[r] += __shfl_xor(lsum[r], off, 16);
#pragma unroll
        for (int r = 0; r < 8; ++r)
            lrun[r] = lrun[r] * alpha[r] + lsum[r];
#pragma unroll
        for (int u = 0; u < 4; ++u)
#pragma unroll
            for (int r = 0; r < 8; ++r)
                o[u][r] *= alpha[r];

        // P (C layout) -> per-wave LDS -> A layout (LDS ops in-order within wave)
#pragma unroll
        for (int t = 0; t < 4; ++t)
#pragma unroll
            for (int r = 0; r < 8; ++r)
                Pl[wave][(hi * 8 + r) * 64 + t * 16 + ln] = (_Float16)s[t][r];

        v16h ap[2];
#pragma unroll
        for (int f = 0; f < 2; ++f) {
            const _Float16* pp = &Pl[wave][ln * 64 + f * 32 + hi * 8];
            v8h c0 = *(const v8h*)pp;
            v8h c1 = *(const v8h*)(pp + 16);
            ap[f] = __builtin_shufflevector(c0, c1, 0,1,2,3,4,5,6,7,8,9,10,11,12,13,14,15);
        }

        // O += P @ V  (B-frags read contiguously from transposed V)
#pragma unroll
        for (int u = 0; u < 4; ++u) {
            o[u] = wmma_f16(ap[0], load_bfrag(&Vt[(u * 16) * 64], 64, ln, hi), o[u]);
            o[u] = wmma_f16(ap[1], load_bfrag(&Vt[(u * 16) * 64 + 32], 64, ln, hi), o[u]);
        }
    }

    // ctx is write-once, never re-read by this pipeline -> non-temporal stores
#pragma unroll
    for (int u = 0; u < 4; ++u)
#pragma unroll
        for (int r = 0; r < 8; ++r) {
            int qrow = q0 + hi * 8 + r;
            __builtin_nontemporal_store(
                o[u][r] * (1.0f / lrun[r]),
                ctx + ((size_t)b * SS + qrow) * DD + h * DH + u * 16 + ln);
        }
}

extern "C" void kernel_launch(void* const* d_in, const int* in_sizes, int n_in,
                              void* d_out, int out_size, void* d_ws, size_t ws_size,
                              hipStream_t stream) {
    const float* hidden = (const float*)d_in[0];
    const float* rel    = (const float*)d_in[1];
    const float* Wq     = (const float*)d_in[2];
    const float* bq     = (const float*)d_in[3];
    const float* Wk     = (const float*)d_in[4];
    const float* bk     = (const float*)d_in[5];
    const float* Wv     = (const float*)d_in[6];
    const float* bv     = (const float*)d_in[7];
    const int*   mask   = (const int*)d_in[8];

    char* ws = (char*)d_ws;
    _Float16* Xh  = (_Float16*)(ws);                         // 16 MB
    _Float16* Wqh = (_Float16*)(ws + (16ull << 20));         //  2 MB
    _Float16* Wkh = (_Float16*)(ws + (18ull << 20));
    _Float16* Wvh = (_Float16*)(ws + (20ull << 20));
    _Float16* Qh  = (_Float16*)(ws + (22ull << 20));         // 16 MB each
    _Float16* Kh  = (_Float16*)(ws + (38ull << 20));
    _Float16* Vh  = (_Float16*)(ws + (54ull << 20));

    float* ctx = (float*)d_out;
    float* msc = (float*)d_out + (size_t)BB * SS * DD;

    const int nX = BB * SS * DD;      // 8388608
    const int nW = DD * DD;           // 1048576
    cvt_f32_f16<<<nX / 256, 256, 0, stream>>>(hidden, Xh, nX);
    cvt_f32_f16<<<nW / 256, 256, 0, stream>>>(Wq, Wqh, nW);
    cvt_f32_f16<<<nW / 256, 256, 0, stream>>>(Wk, Wkh, nW);
    cvt_f32_f16<<<nW / 256, 256, 0, stream>>>(Wv, Wvh, nW);

    // Q/K/V projections: [8192,1024] = X @ W^T + b
    dim3 gP((BB * SS) / 128, DD / 128, 1);
    gemm_wmma<<<gP, 256, 0, stream>>>(Xh, Wqh, bq, Qh, nullptr,
                                      BB * SS, DD, DD, 1.0f, 0, 0, 0);
    gemm_wmma<<<gP, 256, 0, stream>>>(Xh, Wkh, bk, Kh, nullptr,
                                      BB * SS, DD, DD, 1.0f, 0, 0, 0);
    gemm_wmma<<<gP, 256, 0, stream>>>(Xh, Wvh, bv, Vh, nullptr,
                                      BB * SS, DD, DD, 1.0f, 0, 0, 0);

    // Flash attention -> ctx
    attn_flash<<<dim3(SS / 128, HH, BB), 256, 0, stream>>>(Qh, Kh, Vh, rel, mask, ctx);

    // mean scores over heads = (Qfull @ Kfull^T) / (H * sqrt(DH)) = /128, batched over B
    dim3 gM(SS / 128, SS / 128, BB);
    gemm_wmma<<<gM, 256, 0, stream>>>(Qh, Kh, nullptr, nullptr, msc,
                                      SS, SS, DD, 1.0f / 128.0f,
                                      (long)SS * DD, (long)SS * DD, (long)SS * SS);
}
